// EquivariantDecoder_87179246174223
// MI455X (gfx1250) — compile-verified, matrix operand order checked by reading, not run on hardware
//
#include <hip/hip_runtime.h>
#include <hip/hip_bf16.h>

// ---------------------------------------------------------------------------
// EquivariantDecoder (PaiNN-style) for gfx1250 / MI455X
//   - node GEMMs: v_wmma_f32_16x16x32_f16 (f16 in, f32 accumulate)
//   - edge messages: VALU + global f32 atomics (scatter-add segment sums)
// ---------------------------------------------------------------------------

typedef __attribute__((ext_vector_type(16))) _Float16 v16h;
typedef __attribute__((ext_vector_type(8)))  float    v8f;

#define FDIM   78
#define FC     234     // 78*3
#define P4     312     // 4*78
#define NRBF   20
#define CUTF   21.0f
#define PI_F   3.14159265358979323846f

// ======================= generic WMMA GEMM ================================
// C[m*ldc + n*cstr] = act( sum_k A[m*lda + k*astr] * W[k*Nout + n] + bias[n] )
// act: 0=none, 1=silu, 2=relu.  Block = 256 thr (8 waves), 128 rows/block.
// Weights transposed into LDS ([Npad][Kpad] f16), A rows staged as f16.
__global__ void gemm_wmma(const float* __restrict__ A, int lda, int astr,
                          const float* __restrict__ W,
                          const float* __restrict__ bias,
                          float* __restrict__ C, int ldc, int cstr,
                          int M, int K, int Nout, int act)
{
    extern __shared__ char smem[];
    const int kSteps = (K + 31) >> 5;
    const int Kpad   = kSteps << 5;
    const int nTiles = (Nout + 15) >> 4;
    const int Npad   = nTiles << 4;

    _Float16* wl = (_Float16*)smem;                                  // [Npad][Kpad]
    _Float16* al = (_Float16*)(smem + (size_t)Npad * Kpad * 2);      // [128][Kpad]

    const int tid = threadIdx.x;
    const int m0  = blockIdx.x * 128;

    // stage transposed weights: wl[n][k] = W[k*Nout + n]  (zero-pad)
    for (int idx = tid; idx < Npad * Kpad; idx += 256) {
        int n = idx / Kpad, k = idx - n * Kpad;
        float v = (n < Nout && k < K) ? W[(size_t)k * Nout + n] : 0.f;
        wl[idx] = (_Float16)v;
    }
    // stage A rows (f32 -> f16, zero-pad)
    for (int idx = tid; idx < 128 * Kpad; idx += 256) {
        int r = idx / Kpad, k = idx - r * Kpad;
        int m = m0 + r;
        float v = (m < M && k < K) ? A[(size_t)m * lda + (size_t)k * astr] : 0.f;
        al[idx] = (_Float16)v;
    }
    __syncthreads();

    const int wave    = tid >> 5;
    const int lane    = tid & 31;
    const int half16  = lane >> 4;   // which K-half of the fragment this lane holds
    const int l16     = lane & 15;
    const int rowBase = wave * 16;   // 16 rows per wave within the 128-row block

    for (int nt = 0; nt < nTiles; ++nt) {
        v8f acc = {};
        for (int ks = 0; ks < kSteps; ++ks) {
            const int kb = ks << 5;
            // A 16x32 f16 fragment (ISA layout): lane m=l16;
            // elems 0..7 -> K = kb + half16*8 + e ; elems 8..15 -> +16
            const _Float16* ap = &al[(rowBase + l16) * Kpad + kb + half16 * 8];
            union { v16h v; uint4 q[2]; } af;
            af.q[0] = *(const uint4*)(ap);
            af.q[1] = *(const uint4*)(ap + 16);
            // B 32x16 fragment mirrors A with N across lanes; wl is [n][k]
            const _Float16* bp = &wl[(nt * 16 + l16) * Kpad + kb + half16 * 8];
            union { v16h v; uint4 q[2]; } bf;
            bf.q[0] = *(const uint4*)(bp);
            bf.q[1] = *(const uint4*)(bp + 16);
            acc = __builtin_amdgcn_wmma_f32_16x16x32_f16(
                false, af.v, false, bf.v, (short)0, acc, false, false);
        }
        // D layout: VGPR r holds M = r + half16*8 (within tile), N = l16
        const int n = nt * 16 + l16;
        if (n < Nout) {
            const float b = bias ? bias[n] : 0.f;
#pragma unroll
            for (int r = 0; r < 8; ++r) {
                int m = m0 + rowBase + half16 * 8 + r;
                if (m < M) {
                    float v = acc[r] + b;
                    if (act == 1)      v = v / (1.f + __expf(-v));   // silu
                    else if (act == 2) v = fmaxf(v, 0.f);            // relu
                    C[(size_t)m * ldc + (size_t)n * cstr] = v;
                }
            }
        }
    }
}

// ======================= edge geometry (layer-invariant) ===================
__global__ void edge_geom(const float* __restrict__ xyz, const int* __restrict__ nbr,
                          int E, float* __restrict__ rbfE,
                          float* __restrict__ unitE, float* __restrict__ envE)
{
    int e = blockIdx.x * blockDim.x + threadIdx.x;
    if (e >= E) return;
    int i = nbr[2 * e], j = nbr[2 * e + 1];
    float rx = xyz[3 * j + 0] - xyz[3 * i + 0];
    float ry = xyz[3 * j + 1] - xyz[3 * i + 1];
    float rz = xyz[3 * j + 2] - xyz[3 * i + 2];
    float d  = sqrtf(rx * rx + ry * ry + rz * rz + 3e-15f);
    float id = 1.f / d;
    unitE[3 * e + 0] = rx * id;
    unitE[3 * e + 1] = ry * id;
    unitE[3 * e + 2] = rz * id;
    float x = PI_F * d / CUTF;
#pragma unroll
    for (int q = 0; q < NRBF; ++q)
        rbfE[(size_t)e * NRBF + q] = sinf((float)(q + 1) * x) * id;
    envE[e] = (d < CUTF) ? 0.5f * (cosf(x) + 1.f) : 0.f;
}

// ======================= edge messages + scatter ===========================
// One thread per (edge, feature). w_s computed on the fly from rbf (rbf_w is
// L2-resident), then inv = phi[j]*w_s; scatter s0 into H and dv into dV.
__global__ void edge_msg(const int* __restrict__ nbr,
                         const float* __restrict__ rbfE,
                         const float* __restrict__ unitE,
                         const float* __restrict__ envE,
                         const float* __restrict__ phi,
                         const float* __restrict__ rbw,  // (NRBF, 4F)
                         const float* __restrict__ rbb,  // (4F)
                         const float* __restrict__ V,    // old V (N,78,3)
                         float* __restrict__ Hacc,       // H accumulator
                         float* __restrict__ dV,         // V delta accumulator
                         int E)
{
    int idx = blockIdx.x * blockDim.x + threadIdx.x;
    if (idx >= E * FDIM) return;
    int e = idx / FDIM, f = idx - e * FDIM;
    int i = nbr[2 * e], j = nbr[2 * e + 1];
    float env = envE[e];

    float r[NRBF];
#pragma unroll
    for (int q = 0; q < NRBF; ++q) r[q] = rbfE[(size_t)e * NRBF + q];

    float s[4];
#pragma unroll
    for (int k = 0; k < 4; ++k) {
        int col = k * FDIM + f;
        float acc = rbb[col];
#pragma unroll
        for (int q = 0; q < NRBF; ++q) acc = fmaf(r[q], rbw[(size_t)q * P4 + col], acc);
        s[k] = acc * env * phi[(size_t)j * P4 + col];
    }

    atomicAdd(&Hacc[(size_t)i * FDIM + f], s[0]);

    const size_t jb = (size_t)j * FC + (size_t)f * 3;
    const size_t ib = (size_t)i * FC + (size_t)f * 3;
    float vjx = V[jb + 0], vjy = V[jb + 1], vjz = V[jb + 2];
    float vix = V[ib + 0], viy = V[ib + 1], viz = V[ib + 2];
    float cx = viy * vjz - viz * vjy;
    float cy = viz * vjx - vix * vjz;
    float cz = vix * vjy - viy * vjx;
    float ux = unitE[3 * e + 0], uy = unitE[3 * e + 1], uz = unitE[3 * e + 2];
    atomicAdd(&dV[ib + 0], s[2] * ux + s[1] * vjx + s[3] * cx);
    atomicAdd(&dV[ib + 1], s[2] * uy + s[1] * vjy + s[3] * cy);
    atomicAdd(&dV[ib + 2], s[2] * uz + s[1] * vjz + s[3] * cz);
}

// ======================= small elementwise kernels =========================
__global__ void init_state(const float* __restrict__ H0, float* __restrict__ Hc,
                           float* __restrict__ V, float* __restrict__ dV, int Nn)
{
    int idx = blockIdx.x * blockDim.x + threadIdx.x;
    if (idx < Nn * FDIM) Hc[idx] = H0[idx];
    if (idx < Nn * FC) { V[idx] = 0.f; dV[idx] = 0.f; }
}

__global__ void apply_dV(float* __restrict__ V, float* __restrict__ dV, int Nn)
{
    int idx = blockIdx.x * blockDim.x + threadIdx.x;
    if (idx >= Nn * FC) return;
    V[idx] += dV[idx];
    dV[idx] = 0.f;            // re-zero for next layer
}

__global__ void build_X(const float* __restrict__ Hc, const float* __restrict__ vv,
                        float* __restrict__ X, int Nn)
{
    int idx = blockIdx.x * blockDim.x + threadIdx.x;
    if (idx >= Nn * FDIM) return;
    int n = idx / FDIM, f = idx - n * FDIM;
    size_t vb = (size_t)n * FC + (size_t)f * 3;
    float a = vv[vb + 0], b = vv[vb + 1], c = vv[vb + 2];
    X[(size_t)n * (2 * FDIM) + f]        = Hc[idx];
    X[(size_t)n * (2 * FDIM) + FDIM + f] = sqrtf(a * a + b * b + c * c + 1e-15f);
}

__global__ void node_update(float* __restrict__ Hc, float* __restrict__ V,
                            const float* __restrict__ uv, const float* __restrict__ vv,
                            const float* __restrict__ aa, int Nn)
{
    int idx = blockIdx.x * blockDim.x + threadIdx.x;
    if (idx >= Nn * FDIM) return;
    int n = idx / FDIM, f = idx - n * FDIM;
    float avv = aa[(size_t)n * FC + f];
    float asv = aa[(size_t)n * FC + FDIM + f];
    float ass = aa[(size_t)n * FC + 2 * FDIM + f];
    size_t vb = (size_t)n * FC + (size_t)f * 3;
    float dot = 0.f;
#pragma unroll
    for (int c = 0; c < 3; ++c) {
        float u = uv[vb + c], w = vv[vb + c];
        dot = fmaf(u, w, dot);
        V[vb + c] += avv * u;
    }
    Hc[idx] += asv * dot + ass;
}

__global__ void sum_relu(const float* __restrict__ V, float* __restrict__ Vs, int Nn)
{
    int idx = blockIdx.x * blockDim.x + threadIdx.x;
    if (idx >= Nn * FDIM) return;
    int n = idx / FDIM, f = idx - n * FDIM;
    size_t vb = (size_t)n * FC + (size_t)f * 3;
    float s = V[vb + 0] + V[vb + 1] + V[vb + 2];
    Vs[idx] = fmaxf(s, 0.f);
}

__global__ void copy_out(const float* __restrict__ src, float* __restrict__ dst, int nElem)
{
    int idx = blockIdx.x * blockDim.x + threadIdx.x;
    if (idx < nElem) dst[idx] = src[idx];
}

// ======================= host orchestration ================================
static inline void launch_gemm(const float* A, int lda, int astr,
                               const float* W, const float* bias,
                               float* C, int ldc, int cstr,
                               int M, int K, int Nout, int act, hipStream_t s)
{
    int kSteps = (K + 31) / 32, Kpad = kSteps * 32;
    int nTiles = (Nout + 15) / 16, Npad = nTiles * 16;
    size_t shmem = (size_t)Npad * Kpad * 2 + (size_t)128 * Kpad * 2;
    int want = (int)shmem > 65536 ? (int)shmem : 65536;
    (void)hipFuncSetAttribute(reinterpret_cast<const void*>(gemm_wmma),
                              hipFuncAttributeMaxDynamicSharedMemorySize, want);
    dim3 grid((M + 127) / 128), block(256);
    hipLaunchKernelGGL(gemm_wmma, grid, block, shmem, s,
                       A, lda, astr, W, bias, C, ldc, cstr, M, K, Nout, act);
}

extern "C" void kernel_launch(void* const* d_in, const int* in_sizes, int n_in,
                              void* d_out, int out_size, void* d_ws, size_t ws_size,
                              hipStream_t stream)
{
    const float* xyz    = (const float*)d_in[0];
    const int*   nbr    = (const int*)  d_in[1];
    const float* H0     = (const float*)d_in[3];
    const float* msg_w1 = (const float*)d_in[4];
    const float* msg_b1 = (const float*)d_in[5];
    const float* msg_w2 = (const float*)d_in[6];
    const float* msg_b2 = (const float*)d_in[7];
    const float* rbf_w  = (const float*)d_in[8];
    const float* rbf_b  = (const float*)d_in[9];
    const float* upd_wu = (const float*)d_in[10];
    const float* upd_wv = (const float*)d_in[11];
    const float* upd_w1 = (const float*)d_in[12];
    const float* upd_b1 = (const float*)d_in[13];
    const float* upd_w2 = (const float*)d_in[14];
    const float* upd_b2 = (const float*)d_in[15];
    const float* dw1    = (const float*)d_in[16];
    const float* db1    = (const float*)d_in[17];
    const float* dw2    = (const float*)d_in[18];
    const float* db2    = (const float*)d_in[19];

    const int Nn = in_sizes[3] / FDIM;   // 20000
    const int E  = in_sizes[1] / 2;      // 320000
    (void)n_in; (void)out_size; (void)ws_size;

    // workspace partition (256B aligned)
    char* base = (char*)d_ws; size_t off = 0;
    auto alloc = [&](size_t nfloat) -> float* {
        float* p = (float*)(base + off);
        off += ((nfloat * 4 + 255) / 256) * 256;
        return p;
    };
    float* Hc   = alloc((size_t)Nn * FDIM);
    float* V    = alloc((size_t)Nn * FC);
    float* dV   = alloc((size_t)Nn * FC);
    float* phi  = alloc((size_t)Nn * P4);
    float* T1   = alloc((size_t)Nn * FDIM);
    float* uv   = alloc((size_t)Nn * FC);
    float* vv   = alloc((size_t)Nn * FC);
    float* X    = alloc((size_t)Nn * 2 * FDIM);
    float* a1   = alloc((size_t)Nn * FDIM);
    float* aa   = alloc((size_t)Nn * FC);
    float* rbfE = alloc((size_t)E * NRBF);
    float* unitE= alloc((size_t)E * 3);
    float* envE = alloc((size_t)E);
    float* Vs   = alloc((size_t)Nn * FDIM);
    float* o1   = alloc((size_t)Nn * 39);

    float* outH = (float*)d_out;                // (N,78)
    float* outO = outH + (size_t)Nn * FDIM;     // (N,13,3) == (N,39)

    const int B = 256;
    init_state<<<(Nn * FC + B - 1) / B, B, 0, stream>>>(H0, Hc, V, dV, Nn);
    edge_geom <<<(E + B - 1) / B,       B, 0, stream>>>(xyz, nbr, E, rbfE, unitE, envE);

    for (int l = 0; l < 3; ++l) {
        // phi = silu(H @ w1 + b1) @ w2 + b2
        launch_gemm(Hc, FDIM, 1, msg_w1 + (size_t)l * FDIM * FDIM, msg_b1 + (size_t)l * FDIM,
                    T1, FDIM, 1, Nn, FDIM, FDIM, 1, stream);
        launch_gemm(T1, FDIM, 1, msg_w2 + (size_t)l * FDIM * P4, msg_b2 + (size_t)l * P4,
                    phi, P4, 1, Nn, FDIM, P4, 0, stream);
        // edge messages -> scatter into Hc and dV (reads OLD V)
        edge_msg<<<((size_t)E * FDIM + B - 1) / B, B, 0, stream>>>(
            nbr, rbfE, unitE, envE, phi,
            rbf_w + (size_t)l * NRBF * P4, rbf_b + (size_t)l * P4,
            V, Hc, dV, E);
        apply_dV<<<(Nn * FC + B - 1) / B, B, 0, stream>>>(V, dV, Nn);
        // u_v / v_v per xyz channel (strided A/C through the same WMMA kernel)
        for (int c = 0; c < 3; ++c) {
            launch_gemm(V + c, FC, 3, upd_wu + (size_t)l * FDIM * FDIM, nullptr,
                        uv + c, FC, 3, Nn, FDIM, FDIM, 0, stream);
            launch_gemm(V + c, FC, 3, upd_wv + (size_t)l * FDIM * FDIM, nullptr,
                        vv + c, FC, 3, Nn, FDIM, FDIM, 0, stream);
        }
        build_X<<<(Nn * FDIM + B - 1) / B, B, 0, stream>>>(Hc, vv, X, Nn);
        launch_gemm(X, 2 * FDIM, 1, upd_w1 + (size_t)l * 2 * FDIM * FDIM,
                    upd_b1 + (size_t)l * FDIM, a1, FDIM, 1, Nn, 2 * FDIM, FDIM, 1, stream);
        launch_gemm(a1, FDIM, 1, upd_w2 + (size_t)l * FDIM * FC,
                    upd_b2 + (size_t)l * FC, aa, FC, 1, Nn, FDIM, FC, 0, stream);
        node_update<<<(Nn * FDIM + B - 1) / B, B, 0, stream>>>(Hc, V, uv, vv, aa, Nn);
    }

    // head: relu(V.sum(-1)) -> relu(@dw1+db1) -> @dw2+db2
    sum_relu<<<(Nn * FDIM + B - 1) / B, B, 0, stream>>>(V, Vs, Nn);
    launch_gemm(Vs, FDIM, 1, dw1, db1, o1, 39, 1, Nn, FDIM, 39, 2, stream);
    launch_gemm(o1, 39, 1, dw2, db2, outO, 39, 1, Nn, 39, 39, 0, stream);
    copy_out<<<(Nn * FDIM + B - 1) / B, B, 0, stream>>>(Hc, outH, Nn * FDIM);
}
